// TransformerBlock_1812476199286
// MI455X (gfx1250) — matrix operand order repaired and checked by reading, not compiled, not run on hardware
//
#include <hip/hip_runtime.h>
#include <hip/hip_bf16.h>

// ---------------- problem constants (match reference) ----------------
#define N_   50000
#define E_   800000
#define C_   128
#define H_   8
#define D_   16
#define C4_  512

typedef __attribute__((ext_vector_type(16))) _Float16 v16h;
typedef __attribute__((ext_vector_type(8)))  _Float16 v8h;
typedef __attribute__((ext_vector_type(8)))  float    v8f;

union AFrag { v16h v; v8h h[2]; };

// A-fragment (16x32 f16, M x K) from LDS row-major tile with row stride ldk (halfs).
// ISA layout: lane(m = l&15, half = l>>4): elems 0..7 -> K = half*8 + i,
// elems 8..15 -> K = 16 + half*8 + i.  Both runs contiguous -> two 16B LDS loads.
__device__ inline v16h load_a_frag_lds(const _Float16* lds, int lane, int kk, int ldk) {
  int m = lane & 15, half = lane >> 4;
  const _Float16* p = lds + m * ldk + kk + half * 8;
  AFrag f;
  f.h[0] = *(const v8h*)(p);
  f.h[1] = *(const v8h*)(p + 16);
  return f.v;
}

// B-fragment from pre-swizzled f16 weights: fragment (cnTile, kk32) stores,
// for each lane, its 16 halfs contiguously (elem i -> K = half*8 + (i&7) + (i>>3)*16, col = cn + n).
// One fragment = two aligned global_load_b128 per lane.
__device__ inline v16h load_b_frag_packed(const _Float16* __restrict__ Wpk,
                                          int nk32, int kk32, int cnTile, int lane) {
  const _Float16* p = Wpk + ((((size_t)cnTile * nk32 + kk32) * 32) + lane) * 16;
  AFrag f;
  f.h[0] = *(const v8h*)(p);
  f.h[1] = *(const v8h*)(p + 8);
  return f.v;
}

__device__ inline void atomicMaxF(float* addr, float val) {
  // ordered-bits trick: positive floats compare as ints, negative as reversed uints
  if (val >= 0.0f) atomicMax((int*)addr, __float_as_int(val));
  else             atomicMin((unsigned int*)addr, __float_as_uint(val));
}

// ---------------- kernel 0: repack f32 weight [K][ncols] into WMMA B-fragment f16 layout ----
// packed index p = ((cnTile*nk32 + kk32)*32 + lane)*16 + i
//   lane = (n | half<<4), element i -> K = kk32*32 + half*8 + (i&7) + (i>>3)*16, col = cnTile*16 + n
__global__ __launch_bounds__(256)
void pack_weights(const float* __restrict__ W, _Float16* __restrict__ Wpk,
                  int K, int ncols) {
  int p = blockIdx.x * 256 + threadIdx.x;
  if (p >= K * ncols) return;
  int i    = p & 15;
  int lane = (p >> 4) & 31;
  int nk32 = K >> 5;
  int kk32 = (p >> 9) % nk32;
  int tile = p / (nk32 << 9);
  int half = lane >> 4, n = lane & 15;
  int kidx = kk32 * 32 + half * 8 + (i & 7) + ((i >> 3) << 4);
  int col  = tile * 16 + n;
  Wpk[p] = (_Float16)W[(size_t)kidx * ncols + col];
}

// ---------------- kernel 1: fused Q/K/V/skip projections (WMMA) ----------------
__global__ __launch_bounds__(256)
void qkv_skip_gemm(const float* __restrict__ x,
                   const _Float16* __restrict__ Wpk,   // 4 packed 128x128 matrices
                   const float* __restrict__ bq, const float* __restrict__ bk,
                   const float* __restrict__ bv, const float* __restrict__ bs,
                   float* __restrict__ q, float* __restrict__ k,
                   float* __restrict__ v, float* __restrict__ attn) {
  __shared__ _Float16 As[16 * C_];
  const int row0 = blockIdx.x * 16;
  const int tid  = threadIdx.x;
  for (int i = tid; i < 16 * C_; i += 256) {
    int r = i >> 7, c = i & 127;
    As[i] = (_Float16)x[(size_t)(row0 + r) * C_ + c];
  }
  __syncthreads();

  const int wave = tid >> 5, lane = tid & 31;
  const int n = lane & 15, mh = lane >> 4;
#pragma unroll
  for (int t = 0; t < 4; ++t) {
    int job = wave * 4 + t;          // 0..31: 4 matrices x 8 col-tiles
    int mat = job >> 3;
    int cnTile = job & 7;
    int cn = cnTile * 16;
    const _Float16* Wm = Wpk + (size_t)mat * (C_ * C_);
    const float* Bm = (mat == 0) ? bq : (mat == 1) ? bk : (mat == 2) ? bv : bs;
    float*       Om = (mat == 0) ? q  : (mat == 1) ? k  : (mat == 2) ? v  : attn;
    v8f c = {};
#pragma unroll
    for (int kk32 = 0; kk32 < C_ / 32; ++kk32) {
      v16h a = load_a_frag_lds(As, lane, kk32 * 32, C_);
      v16h b = load_b_frag_packed(Wm, C_ / 32, kk32, cnTile, lane);
      c = __builtin_amdgcn_wmma_f32_16x16x32_f16(false, a, false, b,
                                                 (short)0, c, false, false);
    }
    float bias = Bm[cn + n];
#pragma unroll
    for (int r = 0; r < 8; ++r)
      Om[(size_t)(row0 + r + 8 * mh) * C_ + cn + n] = c[r] + bias;
  }
}

// ---------------- kernel 2: init segment-softmax state ----------------
__global__ void init_md(float* __restrict__ m, float* __restrict__ denom) {
  int i = blockIdx.x * 256 + threadIdx.x;
  if (i < N_ * H_) { m[i] = -3.0e38f; denom[i] = 0.0f; }
}

// ---------------- kernel 3: per-edge logits + segment max ----------------
__global__ __launch_bounds__(256)
void edge_logits(const float* __restrict__ q, const float* __restrict__ k,
                 const int* __restrict__ ei, float* __restrict__ logits,
                 float* __restrict__ m) {
  int idx = blockIdx.x * 256 + threadIdx.x;       // E*H = 6.4M
  if (idx >= E_ * H_) return;
  int e = idx >> 3, h = idx & 7;
  int src = ei[e], dst = ei[E_ + e];
  const float4* qp = (const float4*)(q + (size_t)dst * C_ + h * D_);
  const float4* kp = (const float4*)(k + (size_t)src * C_ + h * D_);
  float acc = 0.0f;
#pragma unroll
  for (int j = 0; j < 4; ++j) {
    float4 a = qp[j], b = kp[j];
    acc += a.x * b.x + a.y * b.y + a.z * b.z + a.w * b.w;
  }
  float lg = acc * 0.25f;                          // 1/sqrt(D), D=16
  logits[idx] = lg;
  atomicMaxF(m + (size_t)dst * H_ + h, lg);
}

// ---------------- kernel 4: exp + segment sum ----------------
__global__ __launch_bounds__(256)
void edge_exp(const int* __restrict__ ei, float* __restrict__ logits,
              const float* __restrict__ m, float* __restrict__ denom) {
  int idx = blockIdx.x * 256 + threadIdx.x;
  if (idx >= E_ * H_) return;
  int e = idx >> 3, h = idx & 7;
  int dst = ei[E_ + e];
  float ex = __expf(logits[idx] - m[(size_t)dst * H_ + h]);
  logits[idx] = ex;
  atomicAdd(denom + (size_t)dst * H_ + h, ex);
}

// ---------------- kernel 5: alpha-weighted value scatter ----------------
__global__ __launch_bounds__(256)
void edge_scatter(const int* __restrict__ ei, const float* __restrict__ logits,
                  const float* __restrict__ denom, const float* __restrict__ v,
                  float* __restrict__ attn) {
  int idx = blockIdx.x * 256 + threadIdx.x;       // E * 32 (8 heads x 4 chunks)
  if (idx >= E_ * 32) return;
  int e = idx >> 5;
  int r = idx & 31;
  int h = r >> 2, c4 = (r & 3) * 4;
  int src = ei[e], dst = ei[E_ + e];
  float alpha = logits[(size_t)e * H_ + h] /
                (denom[(size_t)dst * H_ + h] + 1e-16f);
  float4 vv = *(const float4*)(v + (size_t)src * C_ + h * D_ + c4);
  float* o = attn + (size_t)dst * C_ + h * D_ + c4;
  atomicAdd(o + 0, vv.x * alpha);
  atomicAdd(o + 1, vv.y * alpha);
  atomicAdd(o + 2, vv.z * alpha);
  atomicAdd(o + 3, vv.w * alpha);
}

// ---------------- kernels 6/8: out = LN(a + b)*g + beta (one wave per row) ----------------
__global__ __launch_bounds__(256)
void add_ln(const float* __restrict__ a, const float* __restrict__ b,
            const float* __restrict__ g, const float* __restrict__ beta,
            float* __restrict__ out) {
  int wave = threadIdx.x >> 5, lane = threadIdx.x & 31;
  int row = blockIdx.x * 8 + wave;
  if (row >= N_) return;
  float vals[4];
  float s = 0.0f;
#pragma unroll
  for (int j = 0; j < 4; ++j) {
    int c = lane + j * 32;
    vals[j] = a[(size_t)row * C_ + c] + b[(size_t)row * C_ + c];
    s += vals[j];
  }
#pragma unroll
  for (int off = 16; off > 0; off >>= 1) s += __shfl_xor(s, off, 32);
  float mu = s * (1.0f / C_);
  float var = 0.0f;
#pragma unroll
  for (int j = 0; j < 4; ++j) { float d = vals[j] - mu; var += d * d; }
#pragma unroll
  for (int off = 16; off > 0; off >>= 1) var += __shfl_xor(var, off, 32);
  float rs = rsqrtf(var * (1.0f / C_) + 1e-5f);
#pragma unroll
  for (int j = 0; j < 4; ++j) {
    int c = lane + j * 32;
    out[(size_t)row * C_ + c] = (vals[j] - mu) * rs * g[c] + beta[c];
  }
}

// ---------------- kernel 7: fused FFN (WMMA), 16x512 intermediate in LDS ----------------
__global__ __launch_bounds__(256)
void ffn_gemm(const float* __restrict__ h,
              const _Float16* __restrict__ W1pk, const float* __restrict__ b1,
              const _Float16* __restrict__ W2pk, const float* __restrict__ b2,
              float* __restrict__ fout) {
  __shared__ _Float16 Hs[16 * C_];    // 4 KB
  __shared__ _Float16 Ts[16 * C4_];   // 16 KB
  const int row0 = blockIdx.x * 16;
  const int tid  = threadIdx.x;
  for (int i = tid; i < 16 * C_; i += 256) {
    int r = i >> 7, c = i & 127;
    Hs[i] = (_Float16)h[(size_t)(row0 + r) * C_ + c];
  }
  __syncthreads();

  const int wave = tid >> 5, lane = tid & 31;
  const int n = lane & 15, mh = lane >> 4;

  // FFN1: 16x512 = 32 col tiles; wave w -> tiles 4w..4w+3; K = 128
#pragma unroll
  for (int t = 0; t < 4; ++t) {
    int cnTile = wave * 4 + t;
    int cn = cnTile * 16;
    v8f c = {};
#pragma unroll
    for (int kk32 = 0; kk32 < C_ / 32; ++kk32) {
      v16h a = load_a_frag_lds(Hs, lane, kk32 * 32, C_);
      v16h b = load_b_frag_packed(W1pk, C_ / 32, kk32, cnTile, lane);
      c = __builtin_amdgcn_wmma_f32_16x16x32_f16(false, a, false, b,
                                                 (short)0, c, false, false);
    }
    float bias = b1[cn + n];
#pragma unroll
    for (int r = 0; r < 8; ++r) {
      float val = c[r] + bias;
      Ts[(size_t)(r + 8 * mh) * C4_ + cn + n] = (_Float16)(val > 0.0f ? val : 0.0f);
    }
  }
  __syncthreads();

  // FFN2: 16x128 = 8 col tiles; wave w -> tile w; K = 512
  {
    int cnTile = wave;
    int cn = cnTile * 16;
    v8f c = {};
#pragma unroll
    for (int kk32 = 0; kk32 < C4_ / 32; ++kk32) {
      v16h a = load_a_frag_lds(Ts, lane, kk32 * 32, C4_);
      v16h b = load_b_frag_packed(W2pk, C4_ / 32, kk32, cnTile, lane);
      c = __builtin_amdgcn_wmma_f32_16x16x32_f16(false, a, false, b,
                                                 (short)0, c, false, false);
    }
    float bias = b2[cn + n];
#pragma unroll
    for (int r = 0; r < 8; ++r)
      fout[(size_t)(row0 + r + 8 * mh) * C_ + cn + n] = c[r] + bias;
  }
}

// ---------------- host launcher ----------------
extern "C" void kernel_launch(void* const* d_in, const int* in_sizes, int n_in,
                              void* d_out, int out_size, void* d_ws, size_t ws_size,
                              hipStream_t stream) {
  const float* x      = (const float*)d_in[0];
  const int*   ei     = (const int*)  d_in[1];
  const float* Wq     = (const float*)d_in[2];
  const float* bq     = (const float*)d_in[3];
  const float* Wk     = (const float*)d_in[4];
  const float* bk     = (const float*)d_in[5];
  const float* Wv     = (const float*)d_in[6];
  const float* bv     = (const float*)d_in[7];
  const float* Wskip  = (const float*)d_in[8];
  const float* bskip  = (const float*)d_in[9];
  const float* g1     = (const float*)d_in[10];
  const float* beta1  = (const float*)d_in[11];
  const float* W1     = (const float*)d_in[12];
  const float* b1     = (const float*)d_in[13];
  const float* W2     = (const float*)d_in[14];
  const float* b2     = (const float*)d_in[15];
  const float* g2     = (const float*)d_in[16];
  const float* beta2  = (const float*)d_in[17];
  float* out = (float*)d_out;

  // ---- workspace layout ----
  // [0, 393216) bytes: packed f16 weights (Wq,Wk,Wv,Wskip: 4*16384 halfs; W1: 65536; W2: 65536)
  // then float buffers.
  _Float16* wqkvs_pk = (_Float16*)d_ws;                    // 4 * 128*128
  _Float16* w1_pk    = wqkvs_pk + 4 * C_ * C_;             // 128*512
  _Float16* w2_pk    = w1_pk    + C_ * C4_;                // 512*128
  float* ws     = (float*)(w2_pk + C4_ * C_);
  float* qbuf   = ws;                               // N*C  (later: h)
  float* kbuf   = qbuf  + (size_t)N_ * C_;          // N*C  (later: ffn out)
  float* vbuf   = kbuf  + (size_t)N_ * C_;          // N*C
  float* attn   = vbuf  + (size_t)N_ * C_;          // N*C  (skip proj + scatter)
  float* logits = attn  + (size_t)N_ * C_;          // E*H  (later: exp values)
  float* mbuf   = logits + (size_t)E_ * H_;         // N*H
  float* dbuf   = mbuf  + (size_t)N_ * H_;          // N*H
  float* hbuf   = qbuf;
  float* fbuf   = kbuf;

  // 0) repack weights to f16 WMMA B-fragment layout (tiny; L2-resident thereafter)
  pack_weights<<<(C_ * C_  + 255) / 256, 256, 0, stream>>>(Wq,    wqkvs_pk + 0 * C_ * C_, C_,  C_);
  pack_weights<<<(C_ * C_  + 255) / 256, 256, 0, stream>>>(Wk,    wqkvs_pk + 1 * C_ * C_, C_,  C_);
  pack_weights<<<(C_ * C_  + 255) / 256, 256, 0, stream>>>(Wv,    wqkvs_pk + 2 * C_ * C_, C_,  C_);
  pack_weights<<<(C_ * C_  + 255) / 256, 256, 0, stream>>>(Wskip, wqkvs_pk + 3 * C_ * C_, C_,  C_);
  pack_weights<<<(C_ * C4_ + 255) / 256, 256, 0, stream>>>(W1,    w1_pk,                  C_,  C4_);
  pack_weights<<<(C4_ * C_ + 255) / 256, 256, 0, stream>>>(W2,    w2_pk,                  C4_, C_);

  // 1) softmax-state init
  init_md<<<(N_ * H_ + 255) / 256, 256, 0, stream>>>(mbuf, dbuf);
  // 2) fused Q/K/V/skip projections (skip written straight into attn accumulator)
  qkv_skip_gemm<<<N_ / 16, 256, 0, stream>>>(x, wqkvs_pk, bq, bk, bv, bskip,
                                             qbuf, kbuf, vbuf, attn);
  // 3) edge logits + segment max
  edge_logits<<<(E_ * H_ + 255) / 256, 256, 0, stream>>>(qbuf, kbuf, ei, logits, mbuf);
  // 4) exp + segment sum
  edge_exp<<<(E_ * H_ + 255) / 256, 256, 0, stream>>>(ei, logits, mbuf, dbuf);
  // 5) weighted scatter-aggregate of values into attn
  edge_scatter<<<(E_ * 32 + 255) / 256, 256, 0, stream>>>(ei, logits, dbuf, vbuf, attn);
  // 6) h = LN(x + attn)  (q dead -> reuse as h)
  add_ln<<<(N_ + 7) / 8, 256, 0, stream>>>(x, attn, g1, beta1, hbuf);
  // 7) f = relu(h@W1+b1)@W2 + b2  (k dead -> reuse as f)
  ffn_gemm<<<N_ / 16, 256, 0, stream>>>(hbuf, w1_pk, b1, w2_pk, b2, fbuf);
  // 8) out = LN(h + f)
  add_ln<<<(N_ + 7) / 8, 256, 0, stream>>>(hbuf, fbuf, g2, beta2, out);
}
